// SimpleAndEffective_11922829214357
// MI455X (gfx1250) — matrix operand order repaired and checked by reading, not compiled, not run on hardware
//
#include <hip/hip_runtime.h>

// Flip to 0 if the inline asm async-copy path fails to assemble.
#define USE_ASYNC_COPY 1

// ---------------- model constants ----------------
#define BEFF 64      // BATCH*OPT*K
#define SEQL 256     // L
#define VECD 256     // VEC
#define HIDN 256     // HID
#define G3   768     // 3*HID
#define HH2  512     // 2*HID
#define HH8  2048    // 8*HID

typedef __attribute__((ext_vector_type(16))) __bf16 v16bf;
typedef __attribute__((ext_vector_type(8)))  float  v8f;

union Frag16 {
  v16bf v;
  unsigned int d[8];
  uint4 q[2];
};

__device__ __forceinline__ unsigned short f2bf(float f) {
  unsigned int x = __float_as_uint(f);
  x += 0x7FFFu + ((x >> 16) & 1u);
  return (unsigned short)(x >> 16);
}
__device__ __forceinline__ float bf2f(unsigned short u) {
  return __uint_as_float(((unsigned int)u) << 16);
}

// ---- CDNA5 async global->LDS copy (ASYNCcnt path) ----
__device__ __forceinline__ void async_copy_b128(void* lds_ptr, const void* gptr) {
#if USE_ASYNC_COPY
  // LDS byte offset = low 32 bits of the generic pointer (LDS aperture rule).
  unsigned lds_off = (unsigned)(unsigned long long)lds_ptr;
  asm volatile("global_load_async_to_lds_b128 %0, %1, off"
               :: "v"(lds_off), "v"(gptr) : "memory");
#else
  *(uint4*)lds_ptr = *(const uint4*)gptr;
#endif
}
__device__ __forceinline__ void async_copy_wait() {
#if USE_ASYNC_COPY
  asm volatile("s_wait_asynccnt 0x0" ::: "memory");
#endif
}

// ---------------- elementwise helpers ----------------
__global__ void cvt_bf16_kernel(const float* __restrict__ in,
                                unsigned short* __restrict__ out, long n) {
  long i = (long)blockIdx.x * blockDim.x + threadIdx.x;
  if (i < n) out[i] = f2bf(in[i]);
}

__global__ void mulvec_bf16_kernel(const float* __restrict__ x,
                                   const float* __restrict__ w,
                                   unsigned short* __restrict__ out,
                                   int mask, long n) {
  long i = (long)blockIdx.x * blockDim.x + threadIdx.x;
  if (i < n) out[i] = f2bf(x[i] * w[i & mask]);
}

__global__ void rowdot_kernel(const float* __restrict__ x,
                              const float* __restrict__ w,
                              float* __restrict__ out, int D, long rows) {
  long r = (long)blockIdx.x * blockDim.x + threadIdx.x;
  if (r >= rows) return;
  const float* xr = x + r * (long)D;
  float acc = 0.f;
  for (int d = 0; d < D; ++d) acc += xr[d] * w[d];
  out[r] = acc;
}

// ---------------- embedding gather (fp32 -> bf16) ----------------
__global__ void embed_gather_kernel(const int* __restrict__ question,
                                    const int* __restrict__ article,
                                    const float* __restrict__ emb,
                                    unsigned short* __restrict__ qx,
                                    unsigned short* __restrict__ ax) {
  int rt = blockIdx.x;            // 0 .. BEFF*SEQL-1
  int r = rt >> 8, t = rt & 255;
  int d = threadIdx.x;            // 0..255
  int b = r >> 3;                 // OPT*K = 8
  int o = (r >> 1) & 3;           // K = 2
  int qid = question[(b * 4 + o) * SEQL + t];
  int aid = article[(long)r * SEQL + t];
  qx[(long)rt * VECD + d] = f2bf(emb[(long)qid * VECD + d]);
  ax[(long)rt * VECD + d] = f2bf(emb[(long)aid * VECD + d]);
}

// ---------------- generic batched bf16 GEMM (WMMA, double-buffered LDS) ----
// C[b][M,N] (f32) = A[b][M,K] (bf16, row-major, lda) * B
//   btMode=1 : B given as BT [N,K] row-major (ldb = row stride in K units)
//   btMode=0 : B given as [K,N] row-major (ldb = row stride in N units)
// Requires M%128==0, N%64==0, K%32==0.
__global__ __launch_bounds__(256)
void gemm_bf16_kernel(const unsigned short* __restrict__ A,
                      const unsigned short* __restrict__ B,
                      float* __restrict__ C,
                      int M, int N, int K,
                      long sA, long sB, long sC,
                      int lda, int ldb, int ldc, int btMode) {
  __shared__ __align__(16) unsigned short lA[2][128 * 32];  // 16 KB
  __shared__ __align__(16) unsigned short lB[2][64 * 32];   //  8 KB, [n][k]
  const int bz = blockIdx.z;
  A += (long)bz * sA; B += (long)bz * sB; C += (long)bz * sC;
  const int m0 = blockIdx.x * 128, n0 = blockIdx.y * 64;
  const int tid = threadIdx.x;
  const int lane = tid & 31, w = tid >> 5;

  v8f acc[4];
#pragma unroll
  for (int i = 0; i < 4; ++i) acc[i] = (v8f){0, 0, 0, 0, 0, 0, 0, 0};

  // staging geometry (per thread)
  const int liA = tid * 16;                 // A: 128x32, 16 elems/thread
  const int rA = liA >> 5, cA = liA & 31;
  const int liB = tid * 8;                  // B: 64x32 (or 32x64), 8 elems/thread
  const int rBt = liB >> 5, cBt = liB & 31; // BT mode: n,k
  const int rB = liB >> 6, cB = liB & 63;   // KN mode: k,n

  auto stage = [&](int buf, int k0) {
    const unsigned short* aSrc = A + (long)(m0 + rA) * lda + k0 + cA;
    async_copy_b128(&lA[buf][liA], aSrc);
    async_copy_b128(&lA[buf][liA + 8], aSrc + 8);
    __builtin_prefetch(aSrc + 64, 0, 1);    // speculative, next+1 chunk
    if (btMode) {
      const unsigned short* bSrc = B + (long)(n0 + rBt) * ldb + k0 + cBt;
      async_copy_b128(&lB[buf][liB], bSrc);
      __builtin_prefetch(bSrc + 64, 0, 1);
    } else {  // transpose [K,N] -> n-major LDS through registers
      uint4 vv = *(const uint4*)(B + (long)(k0 + rB) * ldb + n0 + cB);
      const unsigned short* vs = (const unsigned short*)&vv;
#pragma unroll
      for (int i = 0; i < 8; ++i) lB[buf][(cB + i) * 32 + rB] = vs[i];
    }
  };

  const int nk = K >> 5;
  stage(0, 0);
  async_copy_wait();
  __syncthreads();

  for (int kc = 0; kc < nk; ++kc) {
    const int cur = kc & 1;
    if (kc + 1 < nk) stage(cur ^ 1, (kc + 1) << 5);

    Frag16 fa;
    {
      const unsigned int* dA = (const unsigned int*)lA[cur];
      int base = (w * 16 + (lane & 15)) * 16;   // 16 dwords per LDS row
      int half = lane >> 4;
      fa.q[0] = *(const uint4*)&dA[base + half * 4];
      fa.q[1] = *(const uint4*)&dA[base + 8 + half * 4];
    }
    {
      const unsigned int* dB = (const unsigned int*)lB[cur];
      int nb = lane & 15, kh = lane >> 4;
#pragma unroll
      for (int nt = 0; nt < 4; ++nt) {
        Frag16 fb;
        int rb = (nt * 16 + nb) * 16 + kh * 4;
        fb.q[0] = *(const uint4*)&dB[rb];
        fb.q[1] = *(const uint4*)&dB[rb + 8];
        acc[nt] = __builtin_amdgcn_wmma_f32_16x16x32_bf16(
            false, fa.v, false, fb.v, (short)0, acc[nt], false, false);
      }
    }
    async_copy_wait();   // next tile landed in LDS
    __syncthreads();     // single barrier per K-chunk (double buffer)
  }

  {
    int mbase = m0 + w * 16 + (lane >> 4) * 8;
    int nb = lane & 15;
#pragma unroll
    for (int nt = 0; nt < 4; ++nt) {
      int n = n0 + nt * 16 + nb;
#pragma unroll
      for (int r = 0; r < 8; ++r)
        C[(long)(mbase + r) * ldc + n] = acc[nt][r];
    }
  }
}

// ---------------- persistent GRU recurrence ----------------
// One WG owns 16 sequences for all 256 steps.
// gi  : [BEFF, SEQL, G3] f32 precomputed x@Wih^T (bias added here)
// whh : [G3, HIDN] bf16 row-major (this IS the BT layout of h@Whh^T)
// out : [BEFF, SEQL, outLd] f32, written at column outOff
__global__ __launch_bounds__(256)
void gru_kernel(const float* __restrict__ gi,
                const unsigned short* __restrict__ whh,
                const float* __restrict__ bih, const float* __restrict__ bhh,
                float* __restrict__ out, int reverse, int outLd, int outOff) {
  __shared__ __align__(16) unsigned short hbf[16 * HIDN]; // 8 KB
  __shared__ float gh[16 * G3];                            // 48 KB
  const int tid = threadIdx.x, lane = tid & 31, w = tid >> 5;
  const int seq0 = blockIdx.x * 16;

  for (int i = tid; i < 16 * HIDN; i += 256) hbf[i] = 0;
  __syncthreads();

  for (int step = 0; step < SEQL; ++step) {
    int t = reverse ? (SEQL - 1 - step) : step;

    // ---- phase 1: gh[16,768] = h(bf16) @ Whh^T via WMMA ----
    Frag16 fa[8];
    {
      const unsigned int* dH = (const unsigned int*)hbf;
      int m = lane & 15, half = lane >> 4;
      int rowbase = m * 128;                 // 256 bf16 = 128 dwords / row
#pragma unroll
      for (int kc = 0; kc < 8; ++kc) {
        int kd0 = kc * 16;
        fa[kc].q[0] = *(const uint4*)&dH[rowbase + kd0 + half * 4];
        fa[kc].q[1] = *(const uint4*)&dH[rowbase + kd0 + 8 + half * 4];
      }
    }
#pragma unroll 1
    for (int nti = 0; nti < 6; ++nti) {      // 48 n-tiles over 8 waves
      int nt = w * 6 + nti;
      v8f acc = {0, 0, 0, 0, 0, 0, 0, 0};
      const int n = nt * 16 + (lane & 15);
      const int kh = lane >> 4;
#pragma unroll
      for (int kc = 0; kc < 8; ++kc) {
        Frag16 fb;
        const uint4* src = (const uint4*)(whh + (long)n * HIDN + kc * 32 + kh * 8);
        fb.q[0] = src[0];
        fb.q[1] = src[2];
        acc = __builtin_amdgcn_wmma_f32_16x16x32_bf16(false, fa[kc].v, false, fb.v,
                                                      (short)0, acc, false, false);
      }
      int half = lane >> 4;
      int col = nt * 16 + (lane & 15);
#pragma unroll
      for (int r = 0; r < 8; ++r) gh[(r + 8 * half) * G3 + col] = acc[r];
    }
    __syncthreads();

    // ---- phase 2: gates + state update ----
    {
      int col = tid;  // 0..255
      float bir = bih[col], biz = bih[HIDN + col], bin = bih[2 * HIDN + col];
      float bhr = bhh[col], bhz = bhh[HIDN + col], bhn = bhh[2 * HIDN + col];
#pragma unroll 1
      for (int row = 0; row < 16; ++row) {
        long base = ((long)(seq0 + row) * SEQL + t) * G3;
        float gir = gi[base + col] + bir;
        float giz = gi[base + HIDN + col] + biz;
        float gin = gi[base + 2 * HIDN + col] + bin;
        float ghr = gh[row * G3 + col] + bhr;
        float ghz = gh[row * G3 + HIDN + col] + bhz;
        float ghn = gh[row * G3 + 2 * HIDN + col] + bhn;
        float rr = 1.f / (1.f + __expf(-(gir + ghr)));
        float zz = 1.f / (1.f + __expf(-(giz + ghz)));
        float nn = tanhf(gin + rr * ghn);
        float hp = bf2f(hbf[row * HIDN + col]);
        float hn = (1.f - zz) * nn + zz * hp;
        out[((long)(seq0 + row) * SEQL + t) * outLd + outOff + col] = hn;
        hbf[row * HIDN + col] = f2bf(hn);
      }
    }
    __syncthreads();
  }
}

// ---------------- BiDAF softmax over j + row max ----------------
__global__ __launch_bounds__(256)
void s_softmax_kernel(const float* __restrict__ s,
                      const float* __restrict__ cvec,
                      const float* __restrict__ qvec,
                      const float* __restrict__ bias3,
                      unsigned short* __restrict__ aout,
                      float* __restrict__ smax) {
  __shared__ float red[256];
  int bi = blockIdx.x;           // b*256 + i
  int bb = bi >> 8;
  int j = threadIdx.x;
  float bsum = bias3[0] + bias3[1] + bias3[2];
  float v = s[(long)bi * 256 + j] + cvec[bi] + qvec[bb * 256 + j] + bsum;
  red[j] = v; __syncthreads();
  for (int off = 128; off; off >>= 1) {
    if (j < off) red[j] = fmaxf(red[j], red[j + off]);
    __syncthreads();
  }
  float mx = red[0]; __syncthreads();
  float e = __expf(v - mx);
  red[j] = e; __syncthreads();
  for (int off = 128; off; off >>= 1) {
    if (j < off) red[j] += red[j + off];
    __syncthreads();
  }
  float inv = 1.f / red[0];
  aout[(long)bi * 256 + j] = f2bf(e * inv);
  if (j == 0) smax[bi] = mx;
}

// ---------------- bb softmax over i + q2c = bb @ c ----------------
__global__ __launch_bounds__(256)
void bb_q2c_kernel(const float* __restrict__ smax,
                   const float* __restrict__ c,
                   float* __restrict__ q2c) {
  __shared__ float red[256];
  __shared__ float bbv[256];
  int b = blockIdx.x, i = threadIdx.x;
  float v = smax[b * 256 + i];
  red[i] = v; __syncthreads();
  for (int off = 128; off; off >>= 1) {
    if (i < off) red[i] = fmaxf(red[i], red[i + off]);
    __syncthreads();
  }
  float mx = red[0]; __syncthreads();
  float e = __expf(v - mx);
  red[i] = e; __syncthreads();
  for (int off = 128; off; off >>= 1) {
    if (i < off) red[i] += red[i + off];
    __syncthreads();
  }
  bbv[i] = e / red[0];
  __syncthreads();
  for (int d = i; d < HH2; d += 256) {
    float acc = 0.f;
    for (int ii = 0; ii < 256; ++ii)
      acc += bbv[ii] * c[((long)b * 256 + ii) * HH2 + d];
    q2c[b * HH2 + d] = acc;
  }
}

// ------- assemble att = relu(concat(c, c2q, c*c2q, c*q2c)) (opt. accumulate) ----
__global__ void att_assemble_kernel(const float* __restrict__ c,
                                    const float* __restrict__ c2q,
                                    const float* __restrict__ q2c,
                                    float* __restrict__ att,
                                    unsigned short* __restrict__ att_bf,
                                    int accumulate) {
  long idx = (long)blockIdx.x * blockDim.x + threadIdx.x;  // over 64*256*512
  if (idx >= (long)BEFF * SEQL * HH2) return;
  int d = idx & (HH2 - 1);
  long bi = idx >> 9;
  int b = (int)(bi >> 8);
  float cv = c[idx], c2 = c2q[idx], qv = q2c[b * HH2 + d];
  float o0 = fmaxf(cv, 0.f);
  float o1 = fmaxf(c2, 0.f);
  float o2 = fmaxf(cv * c2, 0.f);
  float o3 = fmaxf(cv * qv, 0.f);
  long base = bi * HH8 + d;
  if (accumulate) {
    att[base] += o0;
    att[base + HH2] += o1;
    att[base + 2 * HH2] += o2;
    att[base + 3 * HH2] += o3;
  } else {
    att[base] = o0;
    att[base + HH2] = o1;
    att[base + 2 * HH2] = o2;
    att[base + 3 * HH2] = o3;
    att_bf[base] = f2bf(o0);
    att_bf[base + HH2] = f2bf(o1);
    att_bf[base + 2 * HH2] = f2bf(o2);
    att_bf[base + 3 * HH2] = f2bf(o3);
  }
}

// ---------------- final: max over k copies, dot with rank_w ----------------
__global__ __launch_bounds__(256)
void rank_reduce_kernel(const float* __restrict__ s,
                        const float* __restrict__ rw,
                        const float* __restrict__ rb,
                        float* __restrict__ out) {
  __shared__ float red[256];
  int bo = blockIdx.x;                           // 0..31
  const long ROW = (long)SEQL * HH8;             // 524288
  long r0 = (long)(2 * bo) * ROW;
  long r1 = r0 + ROW;
  float acc = 0.f;
  for (long e = threadIdx.x; e < ROW; e += 256)
    acc += fmaxf(s[r0 + e], s[r1 + e]) * rw[e];
  red[threadIdx.x] = acc; __syncthreads();
  for (int off = 128; off; off >>= 1) {
    if (threadIdx.x < off) red[threadIdx.x] += red[threadIdx.x + off];
    __syncthreads();
  }
  if (threadIdx.x == 0) out[bo] = red[0] + rb[0];
}

// ---------------- host orchestration ----------------
extern "C" void kernel_launch(void* const* d_in, const int* in_sizes, int n_in,
                              void* d_out, int out_size, void* d_ws, size_t ws_size,
                              hipStream_t stream) {
  const int* question = (const int*)d_in[0];
  const int* article  = (const int*)d_in[1];
  const float* emb    = (const float*)d_in[2];
  const float* g1_wih = (const float*)d_in[3];
  const float* g1_whh = (const float*)d_in[4];
  const float* g1_bih = (const float*)d_in[5];
  const float* g1_bhh = (const float*)d_in[6];
  const float* g2_wih = (const float*)d_in[7];
  const float* g2_whh = (const float*)d_in[8];
  const float* g2_bih = (const float*)d_in[9];
  const float* g2_bhh = (const float*)d_in[10];
  const float* b1_w   = (const float*)d_in[11];
  const float* b1_b   = (const float*)d_in[12];
  const float* b2_w   = (const float*)d_in[13];
  const float* b2_b   = (const float*)d_in[14];
  const float* rank_w = (const float*)d_in[15];
  const float* rank_b = (const float*)d_in[16];
  float* out = (float*)d_out;

  char* ws = (char*)d_ws;
  size_t off = 0;
  auto alloc = [&](size_t bytes) -> char* {
    char* p = ws + off;
    off = (off + bytes + 255) & ~(size_t)255;
    return p;
  };

  const long NRT  = (long)BEFF * SEQL;              // 16384 rows
  unsigned short* qx_bf   = (unsigned short*)alloc(NRT * VECD * 2);
  unsigned short* ax_bf   = (unsigned short*)alloc(NRT * VECD * 2);
  unsigned short* w1ih_bf = (unsigned short*)alloc(2L * G3 * VECD * 2);
  unsigned short* w1hh_bf = (unsigned short*)alloc(2L * G3 * HIDN * 2);
  unsigned short* w2ih_bf = (unsigned short*)alloc(2L * G3 * HH8 * 2);
  unsigned short* w2hh_bf = (unsigned short*)alloc(2L * G3 * HIDN * 2);
  float* gi    = (float*)alloc(NRT * G3 * 4);       // reused (also holds s scores)
  float* qh    = (float*)alloc(NRT * HH2 * 4);      // layer1 q hidden; later ax2
  float* ah    = (float*)alloc(NRT * HH2 * 4);      // layer1 a hidden
  unsigned short* qside_bf = (unsigned short*)alloc(NRT * HH2 * 2); // ah_bf / ax2_bf
  unsigned short* cw2_bf   = (unsigned short*)alloc(NRT * HH2 * 2);
  unsigned short* sA_bf    = (unsigned short*)alloc((long)BEFF * 256 * 256 * 2);
  float* smax = (float*)alloc(NRT * 4);
  float* cvec = (float*)alloc(NRT * 4);
  float* qvec = (float*)alloc(NRT * 4);
  float* q2c  = (float*)alloc((long)BEFF * HH2 * 4);
  float* c2q  = (float*)alloc(NRT * HH2 * 4);
  float* att  = (float*)alloc(NRT * HH8 * 4);
  unsigned short* att_bf = (unsigned short*)alloc(NRT * HH8 * 2);
  float* ax2 = qh;                                  // reuse after bidaf-1

  auto cvt = [&](const float* src, unsigned short* dst, long n) {
    cvt_bf16_kernel<<<(unsigned)((n + 255) / 256), 256, 0, stream>>>(src, dst, n);
  };
  auto gemm = [&](const unsigned short* A, const unsigned short* B, float* C,
                  int M, int N, int K, long sA, long sB, long sC,
                  int lda, int ldb, int ldc, int bt, int batch) {
    dim3 g(M / 128, N / 64, batch);
    gemm_bf16_kernel<<<g, 256, 0, stream>>>(A, B, C, M, N, K, sA, sB, sC,
                                            lda, ldb, ldc, bt);
  };

  // --- weight prep (fp32 -> bf16, native layout == BT layout) ---
  cvt(g1_wih, w1ih_bf, 2L * G3 * VECD);
  cvt(g1_whh, w1hh_bf, 2L * G3 * HIDN);
  cvt(g2_wih, w2ih_bf, 2L * G3 * HH8);
  cvt(g2_whh, w2hh_bf, 2L * G3 * HIDN);

  // --- embeddings ---
  embed_gather_kernel<<<(unsigned)NRT, 256, 0, stream>>>(question, article, emb,
                                                         qx_bf, ax_bf);

  // --- layer-1 BiGRU (shared encoder) over qx and ax ---
  for (int dir = 0; dir < 2; ++dir) {
    gemm(qx_bf, w1ih_bf + (long)dir * G3 * VECD, gi,
         (int)NRT, G3, VECD, 0, 0, 0, VECD, VECD, G3, 1, 1);
    gru_kernel<<<BEFF / 16, 256, 0, stream>>>(gi, w1hh_bf + (long)dir * G3 * HIDN,
                                              g1_bih + dir * G3, g1_bhh + dir * G3,
                                              qh, dir, HH2, dir * HIDN);
  }
  for (int dir = 0; dir < 2; ++dir) {
    gemm(ax_bf, w1ih_bf + (long)dir * G3 * VECD, gi,
         (int)NRT, G3, VECD, 0, 0, 0, VECD, VECD, G3, 1, 1);
    gru_kernel<<<BEFF / 16, 256, 0, stream>>>(gi, w1hh_bf + (long)dir * G3 * HIDN,
                                              g1_bih + dir * G3, g1_bhh + dir * G3,
                                              ah, dir, HH2, dir * HIDN);
  }

  // --- BiDAF 1: c = qh, q = ah ---
  long nCQ = NRT * HH2;
  cvt(ah, qside_bf, nCQ);
  mulvec_bf16_kernel<<<(unsigned)((nCQ + 255) / 256), 256, 0, stream>>>(
      qh, b1_w + 2 * HH2, cw2_bf, HH2 - 1, nCQ);
  rowdot_kernel<<<(unsigned)((NRT + 255) / 256), 256, 0, stream>>>(qh, b1_w, cvec, HH2, NRT);
  rowdot_kernel<<<(unsigned)((NRT + 255) / 256), 256, 0, stream>>>(ah, b1_w + HH2, qvec, HH2, NRT);
  // s[b] = cw2[b] @ ah[b]^T   (BT mode)
  gemm(cw2_bf, qside_bf, gi, 256, 256, HH2,
       256L * HH2, 256L * HH2, 256L * 256, HH2, HH2, 256, 1, BEFF);
  s_softmax_kernel<<<(unsigned)NRT, 256, 0, stream>>>(gi, cvec, qvec, b1_b, sA_bf, smax);
  // c2q[b] = softmax(s)[b] @ ah[b]   (K x N mode)
  gemm(sA_bf, qside_bf, c2q, 256, HH2, 256,
       256L * 256, 256L * HH2, 256L * HH2, 256, HH2, HH2, 0, BEFF);
  bb_q2c_kernel<<<BEFF, 256, 0, stream>>>(smax, qh, q2c);
  att_assemble_kernel<<<(unsigned)((nCQ + 255) / 256), 256, 0, stream>>>(
      qh, c2q, q2c, att, att_bf, 0);

  // --- layer-2 BiGRU over att (input dim 2048) ---
  for (int dir = 0; dir < 2; ++dir) {
    gemm(att_bf, w2ih_bf + (long)dir * G3 * HH8, gi,
         (int)NRT, G3, HH8, 0, 0, 0, HH8, HH8, G3, 1, 1);
    gru_kernel<<<BEFF / 16, 256, 0, stream>>>(gi, w2hh_bf + (long)dir * G3 * HIDN,
                                              g2_bih + dir * G3, g2_bhh + dir * G3,
                                              ax2, dir, HH2, dir * HIDN);
  }

  // --- BiDAF 2: c = q = ax2; accumulate relu(bidaf) into att ---
  cvt(ax2, qside_bf, nCQ);
  mulvec_bf16_kernel<<<(unsigned)((nCQ + 255) / 256), 256, 0, stream>>>(
      ax2, b2_w + 2 * HH2, cw2_bf, HH2 - 1, nCQ);
  rowdot_kernel<<<(unsigned)((NRT + 255) / 256), 256, 0, stream>>>(ax2, b2_w, cvec, HH2, NRT);
  rowdot_kernel<<<(unsigned)((NRT + 255) / 256), 256, 0, stream>>>(ax2, b2_w + HH2, qvec, HH2, NRT);
  gemm(cw2_bf, qside_bf, gi, 256, 256, HH2,
       256L * HH2, 256L * HH2, 256L * 256, HH2, HH2, 256, 1, BEFF);
  s_softmax_kernel<<<(unsigned)NRT, 256, 0, stream>>>(gi, cvec, qvec, b2_b, sA_bf, smax);
  gemm(sA_bf, qside_bf, c2q, 256, HH2, 256,
       256L * 256, 256L * HH2, 256L * HH2, 256, HH2, HH2, 0, BEFF);
  bb_q2c_kernel<<<BEFF, 256, 0, stream>>>(smax, ax2, q2c);
  att_assemble_kernel<<<(unsigned)((nCQ + 255) / 256), 256, 0, stream>>>(
      ax2, c2q, q2c, att, (unsigned short*)nullptr, 1);

  // --- final ranking ---
  rank_reduce_kernel<<<32, 256, 0, stream>>>(att, rank_w, rank_b, out);
  (void)in_sizes; (void)n_in; (void)out_size; (void)ws_size;
}
